// PSRNN_Model_36979668418850
// MI455X (gfx1250) — compile-verified, matrix operand order checked by reading, not run on hardware
//
#include <hip/hip_runtime.h>

// ---------------- CDNA5 WMMA types ----------------
typedef __bf16 v16bf __attribute__((ext_vector_type(16)));
typedef __bf16 v8bf  __attribute__((ext_vector_type(8)));
typedef float  v8f   __attribute__((ext_vector_type(8)));

static __device__ __forceinline__ __bf16 f2bf(float x) {
  unsigned u = __float_as_uint(x);
  unsigned r = (u + 0x7FFFu + ((u >> 16) & 1u)) >> 16;
  unsigned short s = (unsigned short)r;
  return __builtin_bit_cast(__bf16, s);
}

// Load a 16x32 bf16 fragment (rows = M or N, cols = K) from row-major [rows x ld].
// Layout per ISA 7.12.2: lane = row%16, half = lane>>4 selects K 0-7 / 8-15 (lo regs)
// and 16-23 / 24-31 (hi regs). Two 16B loads per lane -> global_load_b128.
static __device__ __forceinline__ v16bf load_frag(const __bf16* __restrict__ base,
                                                  int ld, int row0, int k0) {
  int lane = threadIdx.x & 31;
  int m    = lane & 15;
  int half = lane >> 4;
  const __bf16* p = base + (size_t)(row0 + m) * (size_t)ld + (size_t)(k0 + half * 8);
  v8bf lo = *(const v8bf*)(p);
  v8bf hi = *(const v8bf*)(p + 16);
  v16bf f;
#pragma unroll
  for (int i = 0; i < 8; ++i) { f[i] = lo[i]; f[i + 8] = hi[i]; }
  return f;
}

static __device__ __forceinline__ v8f wmma_bf16(v16bf a, v16bf b, v8f c) {
  return __builtin_amdgcn_wmma_f32_16x16x32_bf16(false, a, false, b, (short)0, c,
                                                 false, false);
}

// ---------------- constants ----------------
#define BPTT   32
#define BSZ    32
#define NTOKEN 2000
#define NINP   1024
#define NHID   256
#define KPAD   2048   // NTOKEN padded to multiple of 32
#define MROWS  (BPTT * BSZ)   // 1024

// ---------------- convert / pad kernels ----------------
__global__ void k_cvt(const float* __restrict__ src, __bf16* __restrict__ dst, int n) {
  int i = blockIdx.x * blockDim.x + threadIdx.x;
  if (i < n) dst[i] = f2bf(src[i]);
}

__global__ void k_cvt_pad(const float* __restrict__ src, __bf16* __restrict__ dst,
                          int rows, int scols, int dcols) {
  int i = blockIdx.x * blockDim.x + threadIdx.x;
  int tot = rows * dcols;
  if (i >= tot) return;
  int r = i / dcols, c = i - r * dcols;
  dst[i] = (c < scols) ? f2bf(src[(size_t)r * scols + c]) : f2bf(0.0f);
}

// ---------------- encode: enc = bf16( cos(input @ RBF_w^T + RBF_b) * scale ) ------
// M=1024, N=NINP=1024, K=KPAD. One wave per 32x32 tile.
__global__ void k_gemm_encode(const __bf16* __restrict__ A,   // [1024 x KPAD]
                              const __bf16* __restrict__ B,   // [NINP x KPAD] (=B^T)
                              const float* __restrict__ bias, // RBF_b [NINP]
                              __bf16* __restrict__ enc,       // [1024 x NINP]
                              float scale) {
  int wid = (blockIdx.x * blockDim.x + threadIdx.x) >> 5;
  int wm = wid >> 5, wn = wid & 31;          // 32 x 32 tiles
  int row0 = wm * 32, col0 = wn * 32;
  v8f c00 = {0}, c01 = {0}, c10 = {0}, c11 = {0};
  for (int k0 = 0; k0 < KPAD; k0 += 32) {
    v16bf a0 = load_frag(A, KPAD, row0,      k0);
    v16bf a1 = load_frag(A, KPAD, row0 + 16, k0);
    v16bf b0 = load_frag(B, KPAD, col0,      k0);
    v16bf b1 = load_frag(B, KPAD, col0 + 16, k0);
    c00 = wmma_bf16(a0, b0, c00);
    c01 = wmma_bf16(a0, b1, c01);
    c10 = wmma_bf16(a1, b0, c10);
    c11 = wmma_bf16(a1, b1, c11);
  }
  int lane = threadIdx.x & 31, nl = lane & 15, half = lane >> 4;
#pragma unroll
  for (int v = 0; v < 8; ++v) {
    int m0 = row0 + v + 8 * half, m1 = m0 + 16;
    int n0 = col0 + nl,           n1 = n0 + 16;
    enc[(size_t)m0 * NINP + n0] = f2bf(__cosf(c00[v] + bias[n0]) * scale);
    enc[(size_t)m0 * NINP + n1] = f2bf(__cosf(c01[v] + bias[n1]) * scale);
    enc[(size_t)m1 * NINP + n0] = f2bf(__cosf(c10[v] + bias[n0]) * scale);
    enc[(size_t)m1 * NINP + n1] = f2bf(__cosf(c11[v] + bias[n1]) * scale);
  }
}

// ---------------- obs = enc @ emb_w^T + emb_b (fp32 out) --------------------------
// M=1024, N=NHID=256, K=NINP=1024.
__global__ void k_gemm_obs(const __bf16* __restrict__ A,   // [1024 x NINP]
                           const __bf16* __restrict__ B,   // [NHID x NINP]
                           const float* __restrict__ bias, // emb_b [NHID]
                           float* __restrict__ obs) {      // [1024 x NHID]
  int wid = (blockIdx.x * blockDim.x + threadIdx.x) >> 5;
  int wm = wid >> 3, wn = wid & 7;           // 32 x 8 tiles
  int row0 = wm * 32, col0 = wn * 32;
  v8f c00 = {0}, c01 = {0}, c10 = {0}, c11 = {0};
  for (int k0 = 0; k0 < NINP; k0 += 32) {
    v16bf a0 = load_frag(A, NINP, row0,      k0);
    v16bf a1 = load_frag(A, NINP, row0 + 16, k0);
    v16bf b0 = load_frag(B, NINP, col0,      k0);
    v16bf b1 = load_frag(B, NINP, col0 + 16, k0);
    c00 = wmma_bf16(a0, b0, c00);
    c01 = wmma_bf16(a0, b1, c01);
    c10 = wmma_bf16(a1, b0, c10);
    c11 = wmma_bf16(a1, b1, c11);
  }
  int lane = threadIdx.x & 31, nl = lane & 15, half = lane >> 4;
#pragma unroll
  for (int v = 0; v < 8; ++v) {
    int m0 = row0 + v + 8 * half, m1 = m0 + 16;
    int n0 = col0 + nl,           n1 = n0 + 16;
    obs[(size_t)m0 * NHID + n0] = c00[v] + bias[n0];
    obs[(size_t)m0 * NHID + n1] = c01[v] + bias[n1];
    obs[(size_t)m1 * NHID + n0] = c10[v] + bias[n0];
    obs[(size_t)m1 * NHID + n1] = c11[v] + bias[n1];
  }
}

// ---------------- fused step: bn_un[b,r] = sum_c (b_t @ wfefW^T + wfef_b)[b,r,c] * obs_t[b,c]
// One workgroup (256 thr = 8 waves) per output row r. W-tile [32 x 256] stays in
// WMMA accumulators; contraction against LDS-staged obs_t via wave shuffles.
__global__ void k_step(const __bf16* __restrict__ b_bf,    // [BSZ x NHID]
                       const __bf16* __restrict__ wfef_bf, // [NHID*NHID x NHID]
                       const float* __restrict__ wfef_b,   // [NHID*NHID]
                       const float* __restrict__ obs,      // [1024 x NHID]
                       float* __restrict__ bn_un,          // [BSZ x NHID]
                       int t) {
  __shared__ float s_obs[BSZ][NHID];
  __shared__ float s_part[8][BSZ];
  __shared__ float s_bias[BSZ];
  const int tid = threadIdx.x;
  const int r = blockIdx.x;

#pragma unroll
  for (int i = 0; i < BSZ; ++i)
    s_obs[i][tid] = obs[(size_t)(t * BSZ + i) * NHID + tid];
  if (tid < BSZ) s_bias[tid] = 0.0f;
  __syncthreads();

  { // wfef_b contribution: dot(wfef_b[r*256 : +256], obs_t[b,:])
    int b = tid >> 3, ch = tid & 7;
    float s = 0.0f;
    for (int c = ch * 32; c < ch * 32 + 32; ++c)
      s += wfef_b[(size_t)r * NHID + c] * s_obs[b][c];
    atomicAdd(&s_bias[b], s);
  }

  const int wave = tid >> 5, lane = tid & 31;
  const __bf16* Bb = wfef_bf + (size_t)r * NHID * NHID;  // rows c, cols h
  v8f c00 = {0}, c01 = {0}, c10 = {0}, c11 = {0};
#pragma unroll
  for (int k0 = 0; k0 < NHID; k0 += 32) {
    v16bf a0 = load_frag(b_bf, NHID, 0,  k0);
    v16bf a1 = load_frag(b_bf, NHID, 16, k0);
    v16bf b0 = load_frag(Bb,   NHID, wave * 32,      k0);
    v16bf b1 = load_frag(Bb,   NHID, wave * 32 + 16, k0);
    c00 = wmma_bf16(a0, b0, c00);
    c01 = wmma_bf16(a0, b1, c01);
    c10 = wmma_bf16(a1, b0, c10);
    c11 = wmma_bf16(a1, b1, c11);
  }

  // contract W-tile with obs_t
  float p[BSZ];
#pragma unroll
  for (int i = 0; i < BSZ; ++i) p[i] = 0.0f;
  const int nl = lane & 15, half = lane >> 4;
  const int cA = wave * 32 + nl;        // c index of b0/b1-column frags
  const int cB = cA + 16;
#pragma unroll
  for (int v = 0; v < 8; ++v) {
    int m0 = v + 8 * half, m1 = m0 + 16;
    p[m0] += c00[v] * s_obs[m0][cA];
    p[m0] += c01[v] * s_obs[m0][cB];
    p[m1] += c10[v] * s_obs[m1][cA];
    p[m1] += c11[v] * s_obs[m1][cB];
  }
  // full-wave butterfly reduce: afterwards every lane holds total p[i]
  for (int mask = 1; mask < 32; mask <<= 1) {
#pragma unroll
    for (int i = 0; i < BSZ; ++i) p[i] += __shfl_xor(p[i], mask, 32);
  }
  s_part[wave][lane] = p[lane];
  __syncthreads();

  if (tid < BSZ) {
    float s = s_bias[tid];
#pragma unroll
    for (int w = 0; w < 8; ++w) s += s_part[w][tid];
    bn_un[(size_t)tid * NHID + r] = s;
  }
}

// ---------------- normalize: bn = bn_un / ||bn_un||_2 (per batch row) -------------
__global__ void k_norm(const float* __restrict__ bn_un,  // [BSZ x NHID]
                       __bf16* __restrict__ b_bf,        // [BSZ x NHID] next-step A
                       __bf16* __restrict__ bn_all,      // [1024 x NHID] for decode
                       float* __restrict__ b_final,      // d_out tail
                       int t) {
  int tid = threadIdx.x;
  int b = tid >> 3, ch = tid & 7;
  float s = 0.0f;
  for (int c = ch * 32; c < ch * 32 + 32; ++c) {
    float v = bn_un[(size_t)b * NHID + c];
    s += v * v;
  }
  s += __shfl_xor(s, 1, 32);
  s += __shfl_xor(s, 2, 32);
  s += __shfl_xor(s, 4, 32);
  float inv = rsqrtf(s);
  for (int c = ch * 32; c < ch * 32 + 32; ++c) {
    float v = bn_un[(size_t)b * NHID + c] * inv;
    __bf16 vb = f2bf(v);
    b_bf[(size_t)b * NHID + c] = vb;
    bn_all[(size_t)(t * BSZ + b) * NHID + c] = vb;
    if (t == BPTT - 1) b_final[(size_t)b * NHID + c] = v;
  }
}

// ---------------- decode: out = bn_all @ dec_w^T + dec_b --------------------------
// M=1024, N=NTOKEN=2000, K=NHID=256. One wave per 32x16 tile (125 N-tiles).
__global__ void k_decode(const __bf16* __restrict__ A,   // [1024 x NHID]
                         const __bf16* __restrict__ B,   // [NTOKEN x NHID]
                         const float* __restrict__ bias, // dec_b [NTOKEN]
                         float* __restrict__ out) {      // [1024 x NTOKEN]
  int wid = (blockIdx.x * blockDim.x + threadIdx.x) >> 5;
  int wm = wid / 125, wn = wid % 125;
  int row0 = wm * 32, col0 = wn * 16;
  v8f c0 = {0}, c1 = {0};
#pragma unroll
  for (int k0 = 0; k0 < NHID; k0 += 32) {
    v16bf a0 = load_frag(A, NHID, row0,      k0);
    v16bf a1 = load_frag(A, NHID, row0 + 16, k0);
    v16bf bb = load_frag(B, NHID, col0,      k0);
    c0 = wmma_bf16(a0, bb, c0);
    c1 = wmma_bf16(a1, bb, c1);
  }
  int lane = threadIdx.x & 31, nl = lane & 15, half = lane >> 4;
  int n = col0 + nl;
  float bs = bias[n];
#pragma unroll
  for (int v = 0; v < 8; ++v) {
    int m0 = row0 + v + 8 * half, m1 = m0 + 16;
    out[(size_t)m0 * NTOKEN + n] = c0[v] + bs;
    out[(size_t)m1 * NTOKEN + n] = c1[v] + bs;
  }
}

// ---------------- host ----------------
extern "C" void kernel_launch(void* const* d_in, const int* in_sizes, int n_in,
                              void* d_out, int out_size, void* d_ws, size_t ws_size,
                              hipStream_t stream) {
  const float* in_input = (const float*)d_in[0];   // [32,32,2000]
  const float* in_b     = (const float*)d_in[1];   // [32,256]
  const float* in_rbfw  = (const float*)d_in[2];   // [1024,2000]
  const float* in_rbfb  = (const float*)d_in[3];   // [1024]
  const float* in_embw  = (const float*)d_in[4];   // [256,1024]
  const float* in_embb  = (const float*)d_in[5];   // [256]
  const float* in_wfefw = (const float*)d_in[6];   // [65536,256]
  const float* in_wfefb = (const float*)d_in[7];   // [65536]
  const float* in_decw  = (const float*)d_in[8];   // [2000,256]
  const float* in_decb  = (const float*)d_in[9];   // [2000]

  char* ws = (char*)d_ws;
  size_t off = 0;
  auto alloc = [&](size_t bytes) {
    char* p = ws + off;
    off += (bytes + 255) & ~(size_t)255;
    return p;
  };
  __bf16* in_bf   = (__bf16*)alloc((size_t)MROWS * KPAD * 2);      // 4 MB
  __bf16* rbf_bf  = (__bf16*)alloc((size_t)NINP * KPAD * 2);       // 4 MB
  __bf16* enc_bf  = (__bf16*)alloc((size_t)MROWS * NINP * 2);      // 2 MB
  __bf16* emb_bf  = (__bf16*)alloc((size_t)NHID * NINP * 2);       // 0.5 MB
  float*  obs     = (float*) alloc((size_t)MROWS * NHID * 4);      // 1 MB
  __bf16* wfef_bf = (__bf16*)alloc((size_t)NHID * NHID * NHID * 2);// 33.5 MB
  __bf16* dec_bf  = (__bf16*)alloc((size_t)NTOKEN * NHID * 2);     // 1 MB
  __bf16* b_bf    = (__bf16*)alloc((size_t)BSZ * NHID * 2);
  float*  bn_un   = (float*) alloc((size_t)BSZ * NHID * 4);
  __bf16* bn_all  = (__bf16*)alloc((size_t)MROWS * NHID * 2);      // 0.5 MB

  float* out     = (float*)d_out;                 // [32*32*2000]
  float* b_final = out + (size_t)MROWS * NTOKEN;  // [32*256]

  const int T = 256;
  // weight / activation conversion to bf16
  k_cvt_pad<<<(MROWS * KPAD + T - 1) / T, T, 0, stream>>>(in_input, in_bf, MROWS, NTOKEN, KPAD);
  k_cvt_pad<<<(NINP * KPAD + T - 1) / T, T, 0, stream>>>(in_rbfw, rbf_bf, NINP, NTOKEN, KPAD);
  k_cvt<<<(NHID * NINP + T - 1) / T, T, 0, stream>>>(in_embw, emb_bf, NHID * NINP);
  k_cvt<<<(NHID * NHID * NHID + T - 1) / T, T, 0, stream>>>(in_wfefw, wfef_bf, NHID * NHID * NHID);
  k_cvt<<<(NTOKEN * NHID + T - 1) / T, T, 0, stream>>>(in_decw, dec_bf, NTOKEN * NHID);
  k_cvt<<<(BSZ * NHID + T - 1) / T, T, 0, stream>>>(in_b, b_bf, BSZ * NHID);

  // encode + project
  const float scale = 0.044194173824159216f;  // sqrt(2/1024)
  k_gemm_encode<<<(32 * 32) / 8, T, 0, stream>>>(in_bf, rbf_bf, in_rbfb, enc_bf, scale);
  k_gemm_obs<<<(32 * 8) / 8, T, 0, stream>>>(enc_bf, emb_bf, in_embb, obs);

  // sequential scan
  for (int t = 0; t < BPTT; ++t) {
    k_step<<<NHID, T, 0, stream>>>(b_bf, wfef_bf, in_wfefb, obs, bn_un, t);
    k_norm<<<1, T, 0, stream>>>(bn_un, b_bf, bn_all, b_final, t);
  }

  // batched decode over all (t,b) rows
  k_decode<<<(32 * 125) / 8, T, 0, stream>>>(bn_all, dec_bf, in_decb, out);
}
